// GLABlock_35167192219993
// MI455X (gfx1250) — compile-verified, matrix operand order checked by reading, not run on hardware
//
#include <hip/hip_runtime.h>
#include <hip/hip_bf16.h>

// ---------------------------------------------------------------------------
// GLA block for MI455X (gfx1250): TDM-fed double-buffered bf16 WMMA GEMMs +
// chunked gated-linear-attention scan.
// ---------------------------------------------------------------------------

typedef __bf16 bf16;
typedef __attribute__((ext_vector_type(8)))  bf16  bf16x8;
typedef __attribute__((ext_vector_type(16))) bf16  bf16x16;
typedef __attribute__((ext_vector_type(8)))  float floatx8;
typedef __attribute__((ext_vector_type(4)))  unsigned int u32x4;
typedef __attribute__((ext_vector_type(8)))  int i32x8;
typedef __attribute__((ext_vector_type(4)))  int i32x4;

#define DIMC  1024
#define BB    4
#define TT    2048
#define MROWS (BB*TT)      // 8192
#define DKC   512
#define DVC   1024
#define HKC   128
#define HVC   256
#define CHK   64
#define NCHK  32
#define HVB   64           // v-block per scan workgroup
#define QKVG  3072         // fused q|k|v|g projection width

__device__ __forceinline__ floatx8 zero8() {
  floatx8 z;
#pragma unroll
  for (int i = 0; i < 8; ++i) z[i] = 0.f;
  return z;
}

__device__ __forceinline__ floatx8 wmma_bf16(bf16x16 a, bf16x16 b, floatx8 c) {
  return __builtin_amdgcn_wmma_f32_16x16x32_bf16(false, a, false, b, (short)0, c,
                                                 false, false);
}

// A-fragment (16x32 bf16, M=lane%16): elem i -> K = (i<8?0:16) + 8*hh + i%8
__device__ __forceinline__ bf16x16 frag_a(const bf16* p, int hh) {
  bf16x8 lo = *(const bf16x8*)(p + 8 * hh);
  bf16x8 hi = *(const bf16x8*)(p + 16 + 8 * hh);
  bf16x16 r;
#pragma unroll
  for (int i = 0; i < 8; ++i) { r[i] = lo[i]; r[i + 8] = hi[i]; }
  return r;
}
__device__ __forceinline__ bf16x16 frag_b(const bf16* p, int hh) {
  return frag_a(p, hh);
}

// ---------------------------------------------------------------------------
// Tensor Data Mover: load a 128-row x 32-half 2D tile from a row-major
// [rows x rowlen] bf16 matrix into LDS, padding each 64B row with 16B so the
// LDS image has stride 40 halfs (= our [128][40] tile layout).
// ---------------------------------------------------------------------------
__device__ __forceinline__ void tdm_tile_128x32(const bf16* tile_start,
                                                unsigned lds_off, int rowlen,
                                                int rows) {
  const unsigned long long ga = (unsigned long long)(size_t)tile_start;
  u32x4 g0;
  g0[0] = 1u;                                   // count=1 (valid descriptor)
  g0[1] = lds_off;                              // LDS byte address
  g0[2] = (unsigned)ga;                         // global_addr[31:0]
  g0[3] = (unsigned)((ga >> 32) & 0x1FFFFFFull) // global_addr[56:32]
          | 0x80000000u;                        // type=2 ("image")
  i32x8 g1;
  g1[0] = 0x06D10000;  // data_size=2B, pad_enable, interval=16dw, amount=4dw
  g1[1] = (int)(((unsigned)rowlen & 0xffffu) << 16);          // dim0 lo16
  g1[2] = (int)((((unsigned)rowlen >> 16) & 0xffffu)          // dim0 hi16
                | (((unsigned)rows & 0xffffu) << 16));        // dim1 lo16
  g1[3] = (int)((((unsigned)rows >> 16) & 0xffffu)            // dim1 hi16
                | (32u << 16));                               // tile_dim0=32
  g1[4] = 128;                                                // tile_dim1=128
  g1[5] = rowlen;                                             // stride0 lo32
  g1[6] = 0;
  g1[7] = 0;
  i32x4 z4 = {0, 0, 0, 0};
  i32x8 z8 = {0, 0, 0, 0, 0, 0, 0, 0};
  __builtin_amdgcn_tensor_load_to_lds(g0, g1, z4, z4, z8, 0);
}

// ---------------------------------------------------------------------------
// fp32 [K][N] -> bf16 transposed [N][K] (weight prep, 32x32 LDS tiles)
// ---------------------------------------------------------------------------
__global__ __launch_bounds__(256) void transpose_bf16_kernel(
    const float* __restrict__ in, bf16* __restrict__ out, int K, int N) {
  __shared__ bf16 st[32][33];
  const int k0 = blockIdx.y * 32, n0 = blockIdx.x * 32;
  const int r = threadIdx.x >> 5, c = threadIdx.x & 31;
#pragma unroll
  for (int i = 0; i < 4; ++i)
    st[r + 8 * i][c] = (bf16)in[(size_t)(k0 + r + 8 * i) * N + n0 + c];
  __syncthreads();
#pragma unroll
  for (int i = 0; i < 4; ++i)
    out[(size_t)(n0 + r + 8 * i) * K + k0 + c] = st[c][r + 8 * i];
}

// ---------------------------------------------------------------------------
// LayerNorm (1024 wide) -> bf16
// ---------------------------------------------------------------------------
__global__ __launch_bounds__(256) void layernorm_bf16_kernel(
    const float* __restrict__ x, const float* __restrict__ g,
    const float* __restrict__ b, bf16* __restrict__ out) {
  __shared__ float rs[256], rq[256];
  const int row = blockIdx.x, t = threadIdx.x;
  const float* xr = x + (size_t)row * DIMC;
  float4 vx = *(const float4*)(xr + t * 4);
  rs[t] = vx.x + vx.y + vx.z + vx.w;
  rq[t] = vx.x * vx.x + vx.y * vx.y + vx.z * vx.z + vx.w * vx.w;
  __syncthreads();
  for (int off = 128; off > 0; off >>= 1) {
    if (t < off) { rs[t] += rs[t + off]; rq[t] += rq[t + off]; }
    __syncthreads();
  }
  const float mu   = rs[0] * (1.f / DIMC);
  const float var  = rq[0] * (1.f / DIMC) - mu * mu;
  const float rinv = rsqrtf(var + 1e-5f);
#pragma unroll
  for (int i = 0; i < 4; ++i) {
    int c = t * 4 + i;
    out[(size_t)row * DIMC + c] =
        (bf16)(((&vx.x)[i] - mu) * rinv * g[c] + b[c]);
  }
}

// ---------------------------------------------------------------------------
// TDM-fed double-buffered bf16 WMMA GEMM: C[M,N] = A[M,K] @ Bt[N,K]^T
//   mode 0: Cf = acc (+bias)
//   mode 1: Cf = resid + acc (+bias)
//   mode 2: t = leaky_relu(acc + bias, 0.01); Cb = bf16(t*t)
// 128x128 WG tile, 8 waves (4x2), 32x64 per wave. Wave0 drives the TDM.
// ---------------------------------------------------------------------------
__global__ __launch_bounds__(256) void wmma_gemm_kernel(
    const bf16* __restrict__ A, const bf16* __restrict__ Bt,
    const float* __restrict__ bias, const float* __restrict__ resid,
    float* __restrict__ Cf, bf16* __restrict__ Cb,
    int M, int N, int K, int mode) {
  __shared__ __align__(16) bf16 sA[2][128][40];
  __shared__ __align__(16) bf16 sB[2][128][40];

  const int tid  = threadIdx.x;
  const int wave = tid >> 5;
  const int lane = tid & 31;
  const int rowL = lane & 15;
  const int hh   = lane >> 4;
  const int mw   = wave >> 1;   // 0..3  (M direction, 32 rows each)
  const int nw   = wave & 1;    // 0..1  (N direction, 64 cols each)
  const int n0   = blockIdx.x * 128;
  const int m0   = blockIdx.y * 128;

  floatx8 acc[2][4];
#pragma unroll
  for (int i = 0; i < 2; ++i)
#pragma unroll
    for (int j = 0; j < 4; ++j) acc[i][j] = zero8();

  const int nsteps = K >> 5;

  if (wave == 0) {
    tdm_tile_128x32(A + (size_t)m0 * K, (unsigned)(size_t)&sA[0][0][0], K, M);
    tdm_tile_128x32(Bt + (size_t)n0 * K, (unsigned)(size_t)&sB[0][0][0], K, N);
  }

  for (int i = 0; i < nsteps; ++i) {
    const int buf = i & 1, nbuf = buf ^ 1;
    if (wave == 0) {
      if (i + 1 < nsteps) {
        const int k1 = (i + 1) << 5;
        tdm_tile_128x32(A + (size_t)m0 * K + k1,
                        (unsigned)(size_t)&sA[nbuf][0][0], K, M);
        tdm_tile_128x32(Bt + (size_t)n0 * K + k1,
                        (unsigned)(size_t)&sB[nbuf][0][0], K, N);
        __builtin_amdgcn_s_wait_tensorcnt(2);  // current pair complete
      } else {
        __builtin_amdgcn_s_wait_tensorcnt(0);
      }
    }
    __syncthreads();

    bf16x16 af0 = frag_a(&sA[buf][mw * 32 + rowL][0], hh);
    bf16x16 af1 = frag_a(&sA[buf][mw * 32 + 16 + rowL][0], hh);
#pragma unroll
    for (int nt = 0; nt < 4; ++nt) {
      bf16x16 bfr = frag_b(&sB[buf][nw * 64 + nt * 16 + rowL][0], hh);
      acc[0][nt] = wmma_bf16(af0, bfr, acc[0][nt]);
      acc[1][nt] = wmma_bf16(af1, bfr, acc[1][nt]);
    }
    __syncthreads();
  }

#pragma unroll
  for (int mt = 0; mt < 2; ++mt) {
#pragma unroll
    for (int nt = 0; nt < 4; ++nt) {
      const int n = n0 + nw * 64 + nt * 16 + rowL;
      const float bv = bias ? bias[n] : 0.f;
#pragma unroll
      for (int j = 0; j < 8; ++j) {
        const int m = m0 + mw * 32 + mt * 16 + j + 8 * hh;
        const size_t o = (size_t)m * N + n;
        float vv = acc[mt][nt][j] + bv;
        if (mode == 2) {
          float tl = vv > 0.f ? vv : 0.01f * vv;
          Cb[o] = (bf16)(tl * tl);
        } else {
          if (mode == 1) vv += resid[o];
          Cf[o] = vv;
        }
      }
    }
  }
}

// ---------------------------------------------------------------------------
// Low-rank gate path:  tmp = h @ wgk0   (8192x16, K=1024)
// ---------------------------------------------------------------------------
__global__ __launch_bounds__(256) void lowrank1_kernel(
    const bf16* __restrict__ h, const float* __restrict__ w0,
    float* __restrict__ tmp) {
  __shared__ float red[256];
  const int row = blockIdx.x, t = threadIdx.x;
  float p[16];
#pragma unroll
  for (int n = 0; n < 16; ++n) p[n] = 0.f;
  for (int kk = t; kk < DIMC; kk += 256) {
    float hv = (float)h[(size_t)row * DIMC + kk];
    const float* wr = w0 + (size_t)kk * 16;
#pragma unroll
    for (int n = 0; n < 16; ++n) p[n] += hv * wr[n];
  }
  for (int n = 0; n < 16; ++n) {
    red[t] = p[n];
    __syncthreads();
    for (int off = 128; off > 0; off >>= 1) {
      if (t < off) red[t] += red[t + off];
      __syncthreads();
    }
    if (t == 0) tmp[row * 16 + n] = red[0];
    __syncthreads();
  }
}

// gk = log_sigmoid(tmp @ wgk1 + bgk1) / 16
__global__ __launch_bounds__(256) void lowrank2_kernel(
    const float* __restrict__ tmp, const float* __restrict__ w1,
    const float* __restrict__ b1, float* __restrict__ gkout) {
  const int idx = blockIdx.x * 256 + threadIdx.x;
  const int row = idx >> 9, n = idx & 511;
  float z = b1[n];
#pragma unroll
  for (int l = 0; l < 16; ++l) z += tmp[row * 16 + l] * w1[l * DKC + n];
  float ls = fminf(z, 0.f) - log1pf(__expf(-fabsf(z)));
  gkout[(size_t)row * DKC + n] = ls * (1.f / 16.f);
}

// ---------------------------------------------------------------------------
// Chunked GLA scan. One WG per (b, h, v-block of 64): 4*4*4 = 64 WGs.
// fp32 master state + bf16 mirror so all WMMA operands are b128 LDS loads.
// ---------------------------------------------------------------------------
__global__ __launch_bounds__(256) void gla_scan_kernel(
    const float* __restrict__ qkvg,   // [8192][3072] = q|k|v|g
    const float* __restrict__ gk,     // [8192][512]
    float* __restrict__ o) {          // [8192][1024]
  __shared__ __align__(16) float sSt[HVB][HKC + 4];   // S^T[v][d] fp32 master
  __shared__ __align__(16) bf16 sStb[HVB][HKC + 8];   // bf16 mirror
  __shared__ float sBc[CHK][HKC + 4];
  __shared__ __align__(16) bf16 sQb[CHK][HKC + 8];
  __shared__ __align__(16) bf16 sKb[CHK][HKC + 8];
  __shared__ __align__(16) bf16 sKdt[HKC][CHK + 8];
  __shared__ __align__(16) bf16 sVt[HVB][CHK + 8];
  __shared__ __align__(16) bf16 sAm[CHK][CHK + 8];
  __shared__ float sBlast[HKC];

  const int tid  = threadIdx.x;
  const int wave = tid >> 5;
  const int lane = tid & 31;
  const int rowL = lane & 15;
  const int hh   = lane >> 4;

  const int bidx = blockIdx.x;
  const int b    = bidx >> 4;
  const int h    = (bidx >> 2) & 3;
  const int vb   = bidx & 3;
  const int qcol = h * HKC;               // within qkvg
  const int kcol = DKC + h * HKC;
  const int vcol = 2 * DKC + h * HVC + vb * HVB;
  const int ocol = h * HVC + vb * HVB;    // within o
  const float scale = 0.08838834764831845f;  // 128^-0.5

  for (int i = tid; i < HVB * (HKC + 4); i += 256) ((float*)sSt)[i] = 0.f;
  for (int i = tid; i < HVB * (HKC + 8); i += 256) ((bf16*)sStb)[i] = (bf16)0.f;
  __syncthreads();

  for (int c = 0; c < NCHK; ++c) {
    const size_t r0 = (size_t)b * TT + (size_t)c * CHK;

    // prefetch next chunk's q rows (global_prefetch_b8)
    if (c + 1 < NCHK) {
      const size_t rn = r0 + CHK + (tid & 63);
      __builtin_prefetch(qkvg + rn * QKVG + qcol + ((tid >> 6) << 5), 0, 1);
    }

    // 1) cumulative gate sums per feature column
    if (tid < HKC) {
      const int d = tid;
      float run = 0.f;
      for (int j = 0; j < CHK; ++j) {
        run += gk[(r0 + j) * DKC + qcol + d];
        sBc[j][d] = run;
      }
      sBlast[d] = run;
    }
    __syncthreads();

    // 2) elementwise qb / kb / kd^T / v^T tiles (bf16)
    for (int e = tid; e < CHK * HKC; e += 256) {
      const int j = e >> 7, d = e & 127;
      const float bc = sBc[j][d];
      const float qv = qkvg[(r0 + j) * QKVG + qcol + d];
      const float kv = qkvg[(r0 + j) * QKVG + kcol + d];
      sQb[j][d]  = (bf16)(qv * __expf(bc) * scale);
      sKb[j][d]  = (bf16)(kv * __expf(-bc));
      sKdt[d][j] = (bf16)(kv * __expf(sBlast[d] - bc));
    }
    for (int e = tid; e < CHK * HVB; e += 256) {
      const int j = e >> 6, vv = e & 63;
      sVt[vv][j] = (bf16)(qkvg[(r0 + j) * QKVG + vcol + vv]);
    }
    __syncthreads();

    // 3) A = (qb @ kb^T) * tril mask    [64x64, K=128]
#pragma unroll
    for (int tt = 0; tt < 2; ++tt) {
      const int t16 = wave * 2 + tt;
      const int mi = t16 >> 2, ni = t16 & 3;
      floatx8 acc = zero8();
#pragma unroll
      for (int ks = 0; ks < 4; ++ks) {
        bf16x16 af = frag_a(&sQb[mi * 16 + rowL][ks * 32], hh);
        bf16x16 bf_ = frag_b(&sKb[ni * 16 + rowL][ks * 32], hh);
        acc = wmma_bf16(af, bf_, acc);
      }
#pragma unroll
      for (int j = 0; j < 8; ++j) {
        const int i  = mi * 16 + j + 8 * hh;
        const int jj = ni * 16 + rowL;
        sAm[i][jj] = (jj <= i) ? (bf16)acc[j] : (bf16)0.f;
      }
    }
    __syncthreads();

    // 4) o = A @ v  +  qb @ S          [64x64]
#pragma unroll
    for (int tt = 0; tt < 2; ++tt) {
      const int t16 = wave * 2 + tt;
      const int mi = t16 >> 2, ni = t16 & 3;
      floatx8 acc = zero8();
#pragma unroll
      for (int ks = 0; ks < 2; ++ks) {
        bf16x16 af = frag_a(&sAm[mi * 16 + rowL][ks * 32], hh);
        bf16x16 bf_ = frag_b(&sVt[ni * 16 + rowL][ks * 32], hh);
        acc = wmma_bf16(af, bf_, acc);
      }
#pragma unroll
      for (int ks = 0; ks < 4; ++ks) {
        bf16x16 af = frag_a(&sQb[mi * 16 + rowL][ks * 32], hh);
        bf16x16 bf_ = frag_b(&sStb[ni * 16 + rowL][ks * 32], hh);
        acc = wmma_bf16(af, bf_, acc);
      }
#pragma unroll
      for (int j = 0; j < 8; ++j) {
        const int i  = mi * 16 + j + 8 * hh;
        const int vv = ni * 16 + rowL;
        o[(r0 + i) * DVC + ocol + vv] = acc[j];
      }
    }
    __syncthreads();

    // 5) S = diag(exp(blast)) S + kd^T @ v     [128x64, K=64]
#pragma unroll
    for (int tt = 0; tt < 4; ++tt) {
      const int t32 = wave * 4 + tt;
      const int mi = t32 >> 2, ni = t32 & 3;
      floatx8 acc = zero8();
#pragma unroll
      for (int ks = 0; ks < 2; ++ks) {
        bf16x16 af = frag_a(&sKdt[mi * 16 + rowL][ks * 32], hh);
        bf16x16 bf_ = frag_b(&sVt[ni * 16 + rowL][ks * 32], hh);
        acc = wmma_bf16(af, bf_, acc);
      }
#pragma unroll
      for (int j = 0; j < 8; ++j) {
        const int d  = mi * 16 + j + 8 * hh;
        const int vv = ni * 16 + rowL;
        const float ns = sSt[vv][d] * __expf(sBlast[d]) + acc[j];
        sSt[vv][d]  = ns;
        sStb[vv][d] = (bf16)ns;
      }
    }
    __syncthreads();
  }
}

// ---------------------------------------------------------------------------
// Per-head gated RMS norm:  out = bf16( rmsnorm(o)*gw * g*sigmoid(g) )
// ---------------------------------------------------------------------------
__global__ __launch_bounds__(256) void gated_rmsnorm_kernel(
    const float* __restrict__ o, const float* __restrict__ g, int gstride,
    const float* __restrict__ gw, bf16* __restrict__ out) {
  __shared__ float red[256];
  const int row = blockIdx.x, t = threadIdx.x;
  const int head = t >> 6, l = t & 63;
  const float* orow = o + (size_t)row * DVC;
  float4 ov = *(const float4*)(orow + head * HVC + l * 4);
  red[t] = ov.x * ov.x + ov.y * ov.y + ov.z * ov.z + ov.w * ov.w;
  __syncthreads();
  for (int off = 32; off > 0; off >>= 1) {
    if (l < off) red[t] += red[t + off];
    __syncthreads();
  }
  const float rinv = rsqrtf(red[head << 6] * (1.f / HVC) + 1e-5f);
#pragma unroll
  for (int i = 0; i < 4; ++i) {
    const int vc  = l * 4 + i;
    const int col = head * HVC + vc;
    const float on = (&ov.x)[i] * rinv * gw[vc];
    const float gg = g[(size_t)row * gstride + col];
    const float sg = 1.f / (1.f + __expf(-gg));
    out[(size_t)row * DVC + col] = (bf16)(on * gg * sg);
  }
}

// ---------------------------------------------------------------------------
// Host-side launch sequence
// ---------------------------------------------------------------------------
extern "C" void kernel_launch(void* const* d_in, const int* in_sizes, int n_in,
                              void* d_out, int out_size, void* d_ws,
                              size_t ws_size, hipStream_t stream) {
  const float* x    = (const float*)d_in[0];
  const float* ln1g = (const float*)d_in[1];
  const float* ln1b = (const float*)d_in[2];
  const float* wq   = (const float*)d_in[3];
  const float* wk   = (const float*)d_in[4];
  const float* wv   = (const float*)d_in[5];
  const float* wg   = (const float*)d_in[6];
  const float* wgk0 = (const float*)d_in[7];
  const float* wgk1 = (const float*)d_in[8];
  const float* bgk1 = (const float*)d_in[9];
  const float* gnw  = (const float*)d_in[10];
  const float* wo   = (const float*)d_in[11];
  const float* ln2g = (const float*)d_in[12];
  const float* ln2b = (const float*)d_in[13];
  const float* w1   = (const float*)d_in[14];
  const float* b1   = (const float*)d_in[15];
  const float* w2   = (const float*)d_in[16];
  const float* b2   = (const float*)d_in[17];

  char* ws = (char*)d_ws;
  size_t off = 0;
  auto take = [&](size_t bytes) -> char* {
    char* p = ws + off;
    off += (bytes + 255) & ~(size_t)255;
    return p;
  };

  bf16* h_bf    = (bf16*)take((size_t)MROWS * DIMC * 2);
  bf16* h2_bf   = (bf16*)take((size_t)MROWS * DIMC * 2);
  bf16* og_bf   = (bf16*)take((size_t)MROWS * DVC * 2);
  bf16* msq_bf  = (bf16*)take((size_t)MROWS * 3 * DIMC * 2);
  bf16* qkvg_bt = (bf16*)take((size_t)QKVG * DIMC * 2);     // [3072][1024]
  bf16* wot     = (bf16*)take((size_t)DIMC * DVC * 2);      // [1024][1024]
  bf16* w1t     = (bf16*)take((size_t)3 * DIMC * DIMC * 2); // [3072][1024]
  bf16* w2t     = (bf16*)take((size_t)DIMC * 3 * DIMC * 2); // [1024][3072]
  float* qkvg_ws = (float*)take((size_t)MROWS * QKVG * 4);
  float* gk_ws   = (float*)take((size_t)MROWS * DKC * 4);
  float* o_ws    = (float*)take((size_t)MROWS * DVC * 4);
  float* xmid    = (float*)take((size_t)MROWS * DIMC * 4);
  float* tmp16   = (float*)take((size_t)MROWS * 16 * 4);

  // Weight prep: transpose fp32 [K][N] -> bf16 [N][K]
  auto tr = [&](const float* w, bf16* wt, int K, int N) {
    transpose_bf16_kernel<<<dim3(N / 32, K / 32), 256, 0, stream>>>(w, wt, K, N);
  };
  tr(wq, qkvg_bt, DIMC, DKC);                                // rows 0..511
  tr(wk, qkvg_bt + (size_t)DKC * DIMC, DIMC, DKC);           // rows 512..1023
  tr(wv, qkvg_bt + (size_t)2 * DKC * DIMC, DIMC, DVC);       // rows 1024..2047
  tr(wg, qkvg_bt + (size_t)(2 * DKC + DVC) * DIMC, DIMC, DVC);
  tr(wo, wot, DVC, DIMC);
  tr(w1, w1t, DIMC, 3 * DIMC);
  tr(w2, w2t, 3 * DIMC, DIMC);

  // h = LN1(x)
  layernorm_bf16_kernel<<<MROWS, 256, 0, stream>>>(x, ln1g, ln1b, h_bf);

  const dim3 blk(256);
  // fused q|k|v|g projection: [8192,3072] = h @ Wqkvg
  wmma_gemm_kernel<<<dim3(QKVG / 128, MROWS / 128), blk, 0, stream>>>(
      h_bf, qkvg_bt, nullptr, nullptr, qkvg_ws, nullptr, MROWS, QKVG, DIMC, 0);

  // gk = log_sigmoid(h @ wgk0 @ wgk1 + bgk1) / 16
  lowrank1_kernel<<<MROWS, 256, 0, stream>>>(h_bf, wgk0, tmp16);
  lowrank2_kernel<<<(MROWS * DKC) / 256, 256, 0, stream>>>(tmp16, wgk1, bgk1,
                                                           gk_ws);

  // chunked gated linear attention scan
  gla_scan_kernel<<<64, 256, 0, stream>>>(qkvg_ws, gk_ws, o_ws);

  // gated per-head RMS norm (g lives at column 2048 of qkvg)
  gated_rmsnorm_kernel<<<MROWS, 256, 0, stream>>>(o_ws, qkvg_ws + 2 * DKC + DVC,
                                                  QKVG, gnw, og_bf);

  // x_mid = x + og @ wo
  wmma_gemm_kernel<<<dim3(DIMC / 128, MROWS / 128), blk, 0, stream>>>(
      og_bf, wot, nullptr, x, xmid, nullptr, MROWS, DIMC, DVC, 1);

  // h2 = LN2(x_mid)
  layernorm_bf16_kernel<<<MROWS, 256, 0, stream>>>(xmid, ln2g, ln2b, h2_bf);

  // msq = leaky_relu(h2 @ w1 + b1)^2  (stored bf16)
  wmma_gemm_kernel<<<dim3(3 * DIMC / 128, MROWS / 128), blk, 0, stream>>>(
      h2_bf, w1t, b1, nullptr, nullptr, msq_bf, MROWS, 3 * DIMC, DIMC, 2);

  // out = x_mid + msq @ w2 + b2
  wmma_gemm_kernel<<<dim3(DIMC / 128, MROWS / 128), blk, 0, stream>>>(
      msq_bf, w2t, b2, xmid, (float*)d_out, nullptr, MROWS, DIMC, 3 * DIMC, 1);
}
